// ImprovedPointNet_35656818492219
// MI455X (gfx1250) — matrix operand order repaired and checked by reading, not compiled, not run on hardware
//
#include <hip/hip_runtime.h>

typedef __attribute__((ext_vector_type(16))) _Float16 v16h;
typedef __attribute__((ext_vector_type(8)))  _Float16 v8h;
typedef __attribute__((ext_vector_type(8)))  float    v8f;

#define BN_SCALE 0.9999950f   // 1/sqrt(1+1e-5)

// ---------------- WMMA fragment helpers (CDNA5 16x16x32 f16 layouts) ----------------
// A (16x32 f16): lane l holds row (l&15); halves 0..7 = K (l>=16?8:0)+0..7,
//                halves 8..15 = K 16+(l>=16?8:0)+0..7   (ISA 7.12.2)
__device__ __forceinline__ v16h load_a_frag(const _Float16* __restrict__ row, int hi) {
  const v8h lo = *reinterpret_cast<const v8h*>(row + hi * 8);
  const v8h hs = *reinterpret_cast<const v8h*>(row + 16 + hi * 8);
  v16h a;
#pragma unroll
  for (int i = 0; i < 8; ++i) { a[i] = lo[i]; a[8 + i] = hs[i]; }
  return a;
}

// B (32x16 f16): lane l holds column (l&15); half j<8 = B[(l>=16?8:0)+j][col], j>=8 = B[16+(l>=16?8:0)+j-8][col].
// B[k][n] = W[ocol][k] with W row-major [Cout][Cin]; zero-pad k >= cin.
__device__ __forceinline__ v16h build_b_frag(const float* __restrict__ W, int cin,
                                             int ocol, int kbase, int lane) {
  const float* wrow = W + (size_t)ocol * cin;
  const int khalf = (lane >> 4) * 8;
  v16h bf;
#pragma unroll
  for (int j = 0; j < 16; ++j) {
    const int k = kbase + khalf + (j < 8 ? j : j + 8);
    bf[j] = (_Float16)((k < cin) ? wrow[k] : 0.0f);
  }
  return bf;
}

// ---------------- 1) FPS over 32768 points -> 128 samples, per batch ----------------
__global__ __launch_bounds__(1024) void k_fps1(const float* __restrict__ xyz,
                                               int* __restrict__ outIdx,
                                               float* __restrict__ newXyz) {
  const int b = blockIdx.x, tid = threadIdx.x;
  __shared__ float sval[1024];
  __shared__ int   sidx[1024];
  __shared__ int   sCent[128];
  const float* X = xyz + (size_t)b * 32768 * 3;
  float dist[32];
#pragma unroll
  for (int i = 0; i < 32; ++i) dist[i] = 1e10f;
  int cent = 0;
  for (int it = 0; it < 128; ++it) {
    if (tid == 0) sCent[it] = cent;
    const float cx = X[cent*3+0], cy = X[cent*3+1], cz = X[cent*3+2];
    float bv = -1.0f; int bi = 0;
#pragma unroll
    for (int i = 0; i < 32; ++i) {
      const int p = tid + (i << 10);
      const float dx = X[p*3+0]-cx, dy = X[p*3+1]-cy, dz = X[p*3+2]-cz;
      const float nd = fminf(dist[i], dx*dx + dy*dy + dz*dz);
      dist[i] = nd;
      if (nd > bv) { bv = nd; bi = p; }   // first-index tie-break (p increasing)
    }
    sval[tid] = bv; sidx[tid] = bi;
    __syncthreads();
    for (int s = 512; s > 0; s >>= 1) {
      if (tid < s) {
        const float v2 = sval[tid+s]; const int i2 = sidx[tid+s];
        if (v2 > sval[tid] || (v2 == sval[tid] && i2 < sidx[tid])) { sval[tid] = v2; sidx[tid] = i2; }
      }
      __syncthreads();
    }
    cent = sidx[0];
    __syncthreads();
  }
  if (tid < 128) {
    const int ci = sCent[tid];
    outIdx[b*128 + tid] = ci;
    newXyz[(b*128 + tid)*3 + 0] = X[ci*3+0];
    newXyz[(b*128 + tid)*3 + 1] = X[ci*3+1];
    newXyz[(b*128 + tid)*3 + 2] = X[ci*3+2];
  }
}

// ---------------- 2) Ball query SA1: first 8 in-radius indices, wave per center -----
__global__ __launch_bounds__(256) void k_qb1(const float* __restrict__ xyz,
                                             const float* __restrict__ newXyz1,
                                             int* __restrict__ idx1) {
  const int gw = (blockIdx.x * 256 + threadIdx.x) >> 5;   // 0..2047 = b*128+s
  const int lane = threadIdx.x & 31;
  const int b = gw >> 7;
  const float cx = newXyz1[gw*3+0], cy = newXyz1[gw*3+1], cz = newXyz1[gw*3+2];
  const float* X = xyz + (size_t)b * 32768 * 3;
  int count = 0, first = -1;
  for (int base = 0; base < 32768; base += 32) {
    const int p = base + lane;
    const float dx = X[p*3+0]-cx, dy = X[p*3+1]-cy, dz = X[p*3+2]-cz;
    const bool pred = (dx*dx + dy*dy + dz*dz) <= 0.04f;
    const unsigned m = (unsigned)__builtin_amdgcn_ballot_w32(pred);
    if (m) {
      const int pos = count + __builtin_popcount(m & ((1u << lane) - 1u));
      if (pred && pos < 8) idx1[gw*8 + pos] = p;
      if (first < 0) first = base + __builtin_ctz(m);
      count += __builtin_popcount(m);
      if (count >= 8) break;          // early exit; ~1000 expected in-radius pts
    }
  }
  const int cf = count < 8 ? count : 8;
  if (lane >= cf && lane < 8) idx1[gw*8 + lane] = (first < 0) ? 0 : first;
}

// ---------------- 3) SA1 gather + layer1 (3->32) scalar, emit f16 A-matrix ---------
__global__ __launch_bounds__(256) void k_sa1_l1(const float* __restrict__ xyz,
                                                const float* __restrict__ newXyz1,
                                                const int* __restrict__ idx1,
                                                const float* __restrict__ W,
                                                const float* __restrict__ bb,
                                                const float* __restrict__ g,
                                                const float* __restrict__ bt,
                                                _Float16* __restrict__ A1) {
  const int row = blockIdx.x * 256 + threadIdx.x;   // 0..16383 = ((b*128+s)*8+k)
  const int gidx = row >> 3;
  const int b = gidx >> 7;
  const int p = idx1[row];
  const float* X = xyz + ((size_t)b * 32768 + p) * 3;
  const float x0 = X[0] - newXyz1[gidx*3+0];
  const float x1 = X[1] - newXyz1[gidx*3+1];
  const float x2 = X[2] - newXyz1[gidx*3+2];
#pragma unroll 8
  for (int o = 0; o < 32; ++o) {
    const float acc = bb[o] + W[o*3+0]*x0 + W[o*3+1]*x1 + W[o*3+2]*x2;
    A1[(size_t)row*32 + o] = (_Float16)fmaxf(g[o]*(acc*BN_SCALE) + bt[o], 0.0f);
  }
}

// ---------------- 4) SA1 layer2 (32->64) WMMA + fused max over K=8 -----------------
__global__ __launch_bounds__(256) void k_sa1_l2(const _Float16* __restrict__ A1,
                                                const float* __restrict__ W,
                                                const float* __restrict__ bb,
                                                const float* __restrict__ g,
                                                const float* __restrict__ bt,
                                                float* __restrict__ pts1) {
  const int wid = (blockIdx.x * 256 + threadIdx.x) >> 5;  // mtile 0..1023 (16 rows)
  const int lane = threadIdx.x & 31;
  const int hi = lane >> 4, n = lane & 15;
  const _Float16* arow = A1 + ((size_t)wid * 16 + n) * 32;
  const v16h a = load_a_frag(arow, hi);
  const int gidx = wid * 2 + hi;   // rows 0-7 -> group 2*wid, rows 8-15 -> 2*wid+1
#pragma unroll
  for (int nt = 0; nt < 4; ++nt) {
    const int o = nt * 16 + n;
    const v16h bf = build_b_frag(W, 32, o, 0, lane);
    v8f c = {};
    c = __builtin_amdgcn_wmma_f32_16x16x32_f16(false, a, false, bf, (short)0, c, false, false);
    const float bias = bb[o], gg = g[o], be = bt[o];
    float mx = -1e30f;
#pragma unroll
    for (int j = 0; j < 8; ++j)
      mx = fmaxf(mx, fmaxf(gg * ((c[j] + bias) * BN_SCALE) + be, 0.0f));
    pts1[gidx*64 + o] = mx;
  }
}

// ---------------- 5) FPS2 (128->32) + ball query SA2 fused, per batch --------------
__global__ __launch_bounds__(128) void k_fps2_qb2(const float* __restrict__ xyz1,
                                                  int* __restrict__ outIdx2,
                                                  float* __restrict__ newXyz2,
                                                  int* __restrict__ idx2) {
  const int b = blockIdx.x, tid = threadIdx.x;   // 128 threads
  __shared__ float sx[128*3];
  __shared__ float sval[128];
  __shared__ int   sidx[128];
  __shared__ int   sCent[32];
  const float* X = xyz1 + (size_t)b * 128 * 3;
#pragma unroll
  for (int j = 0; j < 3; ++j) sx[tid*3+j] = X[tid*3+j];
  __syncthreads();
  const float px = sx[tid*3+0], py = sx[tid*3+1], pz = sx[tid*3+2];
  float dist = 1e10f;
  int cent = 0;
  for (int it = 0; it < 32; ++it) {
    if (tid == 0) sCent[it] = cent;
    const float dx = px - sx[cent*3+0], dy = py - sx[cent*3+1], dz = pz - sx[cent*3+2];
    dist = fminf(dist, dx*dx + dy*dy + dz*dz);
    sval[tid] = dist; sidx[tid] = tid;
    __syncthreads();
    for (int s = 64; s > 0; s >>= 1) {
      if (tid < s) {
        const float v2 = sval[tid+s]; const int i2 = sidx[tid+s];
        if (v2 > sval[tid] || (v2 == sval[tid] && i2 < sidx[tid])) { sval[tid] = v2; sidx[tid] = i2; }
      }
      __syncthreads();
    }
    cent = sidx[0];
    __syncthreads();
  }
  if (tid < 32) {
    const int ci = sCent[tid];
    outIdx2[b*32+tid] = ci;
    newXyz2[(b*32+tid)*3+0] = sx[ci*3+0];
    newXyz2[(b*32+tid)*3+1] = sx[ci*3+1];
    newXyz2[(b*32+tid)*3+2] = sx[ci*3+2];
  }
  __syncthreads();
  // ball query: one wave per center (K=16, N=128, r^2=0.16)
  const int wv = tid >> 5, lane = tid & 31;
  for (int s = wv; s < 32; s += 4) {
    const int ci = sCent[s];
    const float cx = sx[ci*3+0], cy = sx[ci*3+1], cz = sx[ci*3+2];
    int count = 0, first = -1;
    int* out = idx2 + ((size_t)b*32 + s) * 16;
#pragma unroll
    for (int base = 0; base < 128; base += 32) {
      const int p = base + lane;
      const float dx = sx[p*3+0]-cx, dy = sx[p*3+1]-cy, dz = sx[p*3+2]-cz;
      const bool pred = (dx*dx + dy*dy + dz*dz) <= 0.16f;
      const unsigned m = (unsigned)__builtin_amdgcn_ballot_w32(pred);
      if (m) {
        const int pos = count + __builtin_popcount(m & ((1u << lane) - 1u));
        if (pred && pos < 16) out[pos] = p;
        if (first < 0) first = base + __builtin_ctz(m);
        count += __builtin_popcount(m);
      }
    }
    const int cf = count < 16 ? count : 16;
    if (lane >= cf && lane < 16) out[lane] = (first < 0) ? 0 : first;
  }
}

// ---------------- 6) SA2 gather: concat(xyz_rel[3], pts1[64]) -> f16 [8192,96] -----
__global__ __launch_bounds__(256) void k_sa2_gather(const float* __restrict__ xyz1,
                                                    const float* __restrict__ pts1,
                                                    const int* __restrict__ idx2,
                                                    const float* __restrict__ newXyz2,
                                                    _Float16* __restrict__ A2) {
  const int row = blockIdx.x * 256 + threadIdx.x;   // 0..8191
  const int gidx = row >> 4;
  const int b = gidx >> 5;
  const int p = idx2[row];
  _Float16* dst = A2 + (size_t)row * 96;
  const float* xs = xyz1 + ((size_t)b * 128 + p) * 3;
#pragma unroll
  for (int j = 0; j < 3; ++j) dst[j] = (_Float16)(xs[j] - newXyz2[gidx*3+j]);
  const float* ps = pts1 + ((size_t)b * 128 + p) * 64;
#pragma unroll 8
  for (int c = 0; c < 64; ++c) dst[3 + c] = (_Float16)ps[c];
#pragma unroll
  for (int c = 67; c < 96; ++c) dst[c] = (_Float16)0.0f;    // K-pad to 96
}

// ---------------- 7) SA2 layer1 (67->64) WMMA, K padded to 96 (3 k-steps) ----------
__global__ __launch_bounds__(256) void k_sa2_l1(const _Float16* __restrict__ A2,
                                                const float* __restrict__ W,
                                                const float* __restrict__ bb,
                                                const float* __restrict__ g,
                                                const float* __restrict__ bt,
                                                _Float16* __restrict__ Amid) {
  const int wid = (blockIdx.x * 256 + threadIdx.x) >> 5;   // 0..511
  const int lane = threadIdx.x & 31;
  const int hi = lane >> 4, n = lane & 15;
  const _Float16* arow = A2 + ((size_t)wid * 16 + n) * 96;
  v16h af[3];
#pragma unroll
  for (int kk = 0; kk < 3; ++kk) af[kk] = load_a_frag(arow + kk * 32, hi);
#pragma unroll
  for (int nt = 0; nt < 4; ++nt) {
    const int o = nt * 16 + n;
    v8f c = {};
#pragma unroll
    for (int kk = 0; kk < 3; ++kk) {
      const v16h bf = build_b_frag(W, 67, o, kk * 32, lane);
      c = __builtin_amdgcn_wmma_f32_16x16x32_f16(false, af[kk], false, bf, (short)0, c, false, false);
    }
    const float bias = bb[o], gg = g[o], be = bt[o];
#pragma unroll
    for (int j = 0; j < 8; ++j) {
      const int r = wid * 16 + j + hi * 8;
      const float v = fmaxf(gg * ((c[j] + bias) * BN_SCALE) + be, 0.0f);
      Amid[(size_t)r * 64 + o] = (_Float16)v;
    }
  }
}

// ---------------- 8) SA2 layer2 (64->128) WMMA + fused max over K=16 ---------------
__global__ __launch_bounds__(256) void k_sa2_l2(const _Float16* __restrict__ Amid,
                                                const float* __restrict__ W,
                                                const float* __restrict__ bb,
                                                const float* __restrict__ g,
                                                const float* __restrict__ bt,
                                                float* __restrict__ pts2) {
  const int wid = (blockIdx.x * 256 + threadIdx.x) >> 5;   // 0..511 = one group
  const int lane = threadIdx.x & 31;
  const int hi = lane >> 4, n = lane & 15;
  const _Float16* arow = Amid + ((size_t)wid * 16 + n) * 64;
  const v16h a0 = load_a_frag(arow, hi);
  const v16h a1 = load_a_frag(arow + 32, hi);
#pragma unroll
  for (int nt = 0; nt < 8; ++nt) {
    const int o = nt * 16 + n;
    v8f c = {};
    c = __builtin_amdgcn_wmma_f32_16x16x32_f16(false, a0, false, build_b_frag(W, 64, o, 0,  lane), (short)0, c, false, false);
    c = __builtin_amdgcn_wmma_f32_16x16x32_f16(false, a1, false, build_b_frag(W, 64, o, 32, lane), (short)0, c, false, false);
    const float bias = bb[o], gg = g[o], be = bt[o];
    float mx = -1e30f;
#pragma unroll
    for (int j = 0; j < 8; ++j)
      mx = fmaxf(mx, fmaxf(gg * ((c[j] + bias) * BN_SCALE) + be, 0.0f));
    // combine rows 0-7 (lanes 0-15) with rows 8-15 (lanes 16-31): SWAPX16 swizzle
    const float other = __uint_as_float(
        (unsigned)__builtin_amdgcn_ds_swizzle((int)__float_as_uint(mx), 0x401F));
    mx = fmaxf(mx, other);
    if (hi == 0) pts2[wid * 128 + o] = mx;
  }
}

// ---------------- 9) Attention + heads + fusion + LN + L2-normalize, per batch -----
__global__ __launch_bounds__(256) void k_final(const float* __restrict__ pts2,
    const float* aW1, const float* ab1, const float* ag1, const float* abt1,
    const float* aW2, const float* ab2,
    const float* gW1, const float* gb1, const float* gg_, const float* gbt,
    const float* gW2, const float* gb2,
    const float* lW1, const float* lb1, const float* lg_, const float* lbt,
    const float* lW2, const float* lb2,
    const float* fW, const float* fb, const float* fg, const float* fbt,
    float* __restrict__ out) {
  const int b = blockIdx.x, tid = threadIdx.x;
  __shared__ float sP[32*128];
  __shared__ float sA1[32*32];
  __shared__ float sA2[32*128];
  __shared__ float sAtt[128];
  __shared__ float sMean[128];
  __shared__ float sHg[256];
  __shared__ float sHl[256];
  __shared__ float sGf[256];
  __shared__ float sLf[256];
  __shared__ float sRed[256];
  const float* P = pts2 + (size_t)b * 4096;
  for (int i = tid; i < 4096; i += 256) sP[i] = P[i];
  __syncthreads();
  for (int t = tid; t < 1024; t += 256) {           // a1 = relu(bn(pts2 @ W1^T))
    const int s = t >> 5, o = t & 31;
    float acc = ab1[o];
    const float* w = aW1 + o * 128;
    const float* x = sP + s * 128;
    for (int c = 0; c < 128; ++c) acc += x[c] * w[c];
    sA1[s*32+o] = fmaxf(ag1[o]*(acc*BN_SCALE) + abt1[o], 0.0f);
  }
  __syncthreads();
  for (int t = tid; t < 4096; t += 256) {           // a2 = a1 @ W2^T + b2
    const int s = t >> 7, o = t & 127;
    float acc = ab2[o];
    const float* w = aW2 + o * 32;
    const float* x = sA1 + s * 32;
    for (int c = 0; c < 32; ++c) acc += x[c] * w[c];
    sA2[s*128+o] = acc;
  }
  __syncthreads();
  if (tid < 128) {                                  // softmax over S + attended + mean
    const int o = tid;
    float mx = -1e30f;
    for (int s = 0; s < 32; ++s) mx = fmaxf(mx, sA2[s*128+o]);
    float sum = 0.0f;
    for (int s = 0; s < 32; ++s) sum += expf(sA2[s*128+o] - mx);
    float att = 0.0f, mn = 0.0f;
    for (int s = 0; s < 32; ++s) {
      att += sP[s*128+o] * (expf(sA2[s*128+o] - mx) / sum);
      mn  += sP[s*128+o];
    }
    sAtt[o] = att; sMean[o] = mn * (1.0f/32.0f);
  }
  __syncthreads();
  {                                                 // head hidden layers
    const int o = tid;
    float aG = gb1[o], aL = lb1[o];
    const float* wg = gW1 + o*128;
    const float* wl = lW1 + o*128;
    for (int c = 0; c < 128; ++c) { aG += sAtt[c]*wg[c]; aL += sMean[c]*wl[c]; }
    sHg[o] = fmaxf(gg_[o]*(aG*BN_SCALE)+gbt[o], 0.0f);
    sHl[o] = fmaxf(lg_[o]*(aL*BN_SCALE)+lbt[o], 0.0f);
  }
  __syncthreads();
  {                                                 // head output layers
    const int o = tid;
    float aG = gb2[o], aL = lb2[o];
    const float* wg = gW2 + o*256;
    const float* wl = lW2 + o*256;
    for (int c = 0; c < 256; ++c) { aG += sHg[c]*wg[c]; aL += sHl[c]*wl[c]; }
    sGf[o] = aG; sLf[o] = aL;
  }
  __syncthreads();
  float f;
  {                                                 // fusion: concat @ fus_W^T + b
    const int o = tid;
    float acc = fb[o];
    const float* w = fW + o*512;
    for (int c = 0; c < 256; ++c) acc += sGf[c]*w[c];
    for (int c = 0; c < 256; ++c) acc += sLf[c]*w[256+c];
    f = acc;
  }
  sRed[tid] = f; __syncthreads();
  for (int s = 128; s > 0; s >>= 1) { if (tid < s) sRed[tid] += sRed[tid+s]; __syncthreads(); }
  const float mu = sRed[0] * (1.0f/256.0f);
  __syncthreads();
  sRed[tid] = f*f; __syncthreads();
  for (int s = 128; s > 0; s >>= 1) { if (tid < s) sRed[tid] += sRed[tid+s]; __syncthreads(); }
  const float var = sRed[0] * (1.0f/256.0f) - mu*mu;
  __syncthreads();
  const float y = fmaxf(fg[tid]*(f-mu)/sqrtf(var + 1e-5f) + fbt[tid], 0.0f);
  sRed[tid] = y*y; __syncthreads();
  for (int s = 128; s > 0; s >>= 1) { if (tid < s) sRed[tid] += sRed[tid+s]; __syncthreads(); }
  const float nrm = sqrtf(sRed[0]);
  out[b*256 + tid] = y / fmaxf(nrm, 1e-12f);
}

extern "C" void kernel_launch(void* const* d_in, const int* in_sizes, int n_in,
                              void* d_out, int out_size, void* d_ws, size_t ws_size,
                              hipStream_t stream) {
  (void)in_sizes; (void)n_in; (void)out_size; (void)ws_size;
  // setup_inputs() dict order: x, then params in insertion order
  const float* x     = (const float*)d_in[0];
  const float* s10W  = (const float*)d_in[1];
  const float* s10b  = (const float*)d_in[2];
  const float* s10g  = (const float*)d_in[3];
  const float* s10bt = (const float*)d_in[4];
  const float* s11W  = (const float*)d_in[5];
  const float* s11b  = (const float*)d_in[6];
  const float* s11g  = (const float*)d_in[7];
  const float* s11bt = (const float*)d_in[8];
  const float* s20W  = (const float*)d_in[9];
  const float* s20b  = (const float*)d_in[10];
  const float* s20g  = (const float*)d_in[11];
  const float* s20bt = (const float*)d_in[12];
  const float* s21W  = (const float*)d_in[13];
  const float* s21b  = (const float*)d_in[14];
  const float* s21g  = (const float*)d_in[15];
  const float* s21bt = (const float*)d_in[16];
  const float* aW1   = (const float*)d_in[17];
  const float* ab1   = (const float*)d_in[18];
  const float* ag1   = (const float*)d_in[19];
  const float* abt1  = (const float*)d_in[20];
  const float* aW2   = (const float*)d_in[21];
  const float* ab2   = (const float*)d_in[22];
  const float* gW1   = (const float*)d_in[23];
  const float* gb1   = (const float*)d_in[24];
  const float* gg    = (const float*)d_in[25];
  const float* gbt   = (const float*)d_in[26];
  const float* gW2   = (const float*)d_in[27];
  const float* gb2   = (const float*)d_in[28];
  const float* lW1   = (const float*)d_in[29];
  const float* lb1   = (const float*)d_in[30];
  const float* lg    = (const float*)d_in[31];
  const float* lbt   = (const float*)d_in[32];
  const float* lW2   = (const float*)d_in[33];
  const float* lb2   = (const float*)d_in[34];
  const float* fW    = (const float*)d_in[35];
  const float* fb    = (const float*)d_in[36];
  const float* fg    = (const float*)d_in[37];
  const float* fbt   = (const float*)d_in[38];

  char* ws = (char*)d_ws;                      // ~4.4 MB total, 16B-aligned slices
  int*      fpsIdx1 = (int*)      (ws + 0);          // 16*128
  float*    newXyz1 = (float*)    (ws + 8192);       // 16*128*3
  int*      idx1    = (int*)      (ws + 32768);      // 16*128*8
  _Float16* A1      = (_Float16*) (ws + 98304);      // 16384 x 32 f16
  float*    pts1    = (float*)    (ws + 1146880);    // 16*128*64
  int*      fpsIdx2 = (int*)      (ws + 1671168);    // 16*32
  float*    newXyz2 = (float*)    (ws + 1673216);    // 16*32*3
  int*      idx2    = (int*)      (ws + 1679360);    // 16*32*16
  _Float16* A2in    = (_Float16*) (ws + 1712128);    // 8192 x 96 f16
  _Float16* A2mid   = (_Float16*) (ws + 3284992);    // 8192 x 64 f16
  float*    pts2    = (float*)    (ws + 4333568);    // 16*32*128

  k_fps1      <<<16, 1024, 0, stream>>>(x, fpsIdx1, newXyz1);
  k_qb1       <<<256, 256, 0, stream>>>(x, newXyz1, idx1);
  k_sa1_l1    <<<64,  256, 0, stream>>>(x, newXyz1, idx1, s10W, s10b, s10g, s10bt, A1);
  k_sa1_l2    <<<128, 256, 0, stream>>>(A1, s11W, s11b, s11g, s11bt, pts1);
  k_fps2_qb2  <<<16,  128, 0, stream>>>(newXyz1, fpsIdx2, newXyz2, idx2);
  k_sa2_gather<<<32,  256, 0, stream>>>(newXyz1, pts1, idx2, newXyz2, A2in);
  k_sa2_l1    <<<64,  256, 0, stream>>>(A2in, s20W, s20b, s20g, s20bt, A2mid);
  k_sa2_l2    <<<64,  256, 0, stream>>>(A2mid, s21W, s21b, s21g, s21bt, pts2);
  k_final     <<<16,  256, 0, stream>>>(pts2,
      aW1, ab1, ag1, abt1, aW2, ab2,
      gW1, gb1, gg, gbt, gW2, gb2,
      lW1, lb1, lg, lbt, lW2, lb2,
      fW, fb, fg, fbt, (float*)d_out);
}
